// SSIGformer_7464653161214
// MI455X (gfx1250) — compile-verified, matrix operand order checked by reading, not run on hardware
//
#include <hip/hip_runtime.h>
#include <hip/hip_bf16.h>

typedef _Float16 f16;
typedef __attribute__((ext_vector_type(16))) _Float16 v16h;
typedef __attribute__((ext_vector_type(8)))  _Float16 v8h;
typedef __attribute__((ext_vector_type(8)))  float    v8f;

#define B_   2
#define N_   2048
#define C_   128
#define H_   4
#define HD_  32
#define C4_  32

// ---------------------------------------------------------------------------
// WMMA fragment helpers (16x16x32 f16 -> f32), wave32, per ISA 7.12.2 layouts.
// A (16x32, MxK) and B (N rows x K) fragments: lane L holds row (L&15);
// halfwords 0..7 at K = base+0..7, 8..15 at K = base+16..23 with base = 0
// (lanes 0-15) or 8 (lanes 16-31). Two 16B contiguous loads per lane.
// ---------------------------------------------------------------------------
__device__ __forceinline__ v16h frag_ld(const f16* __restrict__ base, int ld) {
  int lane = threadIdx.x & 31;
  int r  = lane & 15;
  int kb = (lane & 16) ? 8 : 0;
  const f16* p = base + (long)r * ld + kb;
  v8h lo = *(const v8h*)(p);
  v8h hi = *(const v8h*)(p + 16);
  v16h f;
#pragma unroll
  for (int i = 0; i < 8; ++i) { f[i] = lo[i]; f[i + 8] = hi[i]; }
  return f;
}

__device__ __forceinline__ v8f wmma16(v16h a, v16h b, v8f c) {
  return __builtin_amdgcn_wmma_f32_16x16x32_f16(
      false, a, false, b, (short)0, c, false, false);
}

// Async global->LDS 16B copy (GLOBAL_LOAD_ASYNC_TO_LDS_B128, ASYNCcnt).
__device__ __forceinline__ void async_cp16(unsigned lds_off, const void* gptr) {
  asm volatile("global_load_async_to_lds_b128 %0, %1, off"
               :: "v"(lds_off), "v"(gptr) : "memory");
}
__device__ __forceinline__ void async_wait0() {
  asm volatile("s_wait_asynccnt 0" ::: "memory");
}

// ---------------------------------------------------------------------------
// Plain batched NT GEMM (direct-to-VGPR fragments), 32x32 per wave,
// 4 waves/block arranged wm x wn (wm*wn==4).
// D[b,m,n] = sum_k A[b,m,k]*Bm[b,n,k] (+bias[n]) (+resid[b,m,n]).
// grid = (M/(32*wm), Nn/(32*wn), batch), block = 128.
// ---------------------------------------------------------------------------
__global__ void k_gemm_nt(const f16* __restrict__ A, int lda, long sA,
                          const f16* __restrict__ Bm, int ldb, long sB,
                          const float* __restrict__ bias,
                          const float* __restrict__ resid, long sR,
                          float* __restrict__ Df, f16* __restrict__ Dh,
                          int ldd, long sD, int K, int wm, int wn)
{
  int b  = blockIdx.z;
  int w  = threadIdx.x >> 5;
  int wi = w % wm, wj = w / wm;
  int m0 = blockIdx.x * (32 * wm) + wi * 32;
  int n0 = blockIdx.y * (32 * wn) + wj * 32;
  const f16* Ab = A  + (long)b * sA + (long)m0 * lda;
  const f16* Bb = Bm + (long)b * sB + (long)n0 * ldb;
  v8f a00 = {}, a01 = {}, a10 = {}, a11 = {};
  for (int k = 0; k < K; k += 32) {
    v16h fa0 = frag_ld(Ab + k, lda);
    v16h fa1 = frag_ld(Ab + (long)16 * lda + k, lda);
    v16h fb0 = frag_ld(Bb + k, ldb);
    v16h fb1 = frag_ld(Bb + (long)16 * ldb + k, ldb);
    a00 = wmma16(fa0, fb0, a00);
    a01 = wmma16(fa0, fb1, a01);
    a10 = wmma16(fa1, fb0, a10);
    a11 = wmma16(fa1, fb1, a11);
  }
  int lane = threadIdx.x & 31;
  int n  = lane & 15;
  int mb = (lane & 16) ? 8 : 0;
#pragma unroll
  for (int s = 0; s < 4; ++s) {
    const v8f* accs[4] = {&a00, &a01, &a10, &a11};
    v8f acc = *accs[s];
    int mo = (s >> 1) * 16, no = (s & 1) * 16;
    int col = n0 + no + n;
    float bv = bias ? bias[col] : 0.0f;
#pragma unroll
    for (int r = 0; r < 8; ++r) {
      int row = m0 + mo + mb + r;
      long off = (long)b * sD + (long)row * ldd + col;
      float v = acc[r] + bv;
      if (resid) v += resid[(long)b * sR + (long)row * ldd + col];
      if (Df) Df[off] = v;
      if (Dh) Dh[off] = (f16)v;
    }
  }
}

// ---------------------------------------------------------------------------
// Async-staged NT GEMM: 64x64 block tile, 4 waves (2x2) of 32x32 each.
// A/B k-slabs (64x32 f16) double-buffered in LDS, filled with
// global_load_async_to_lds_b128 and consumed after s_wait_asynccnt+barrier.
// grid = (M/64, Nn/64, batch), block = 128.
// ---------------------------------------------------------------------------
__global__ void k_gemm_nt_async(const f16* __restrict__ A, int lda, long sA,
                                const f16* __restrict__ Bm, int ldb, long sB,
                                const float* __restrict__ bias,
                                const float* __restrict__ resid, long sR,
                                float* __restrict__ Df, f16* __restrict__ Dh,
                                int ldd, long sD, int K)
{
  __shared__ __align__(16) f16 sbuf[2][2][64 * 32]; // [buf][A/B][row*32+k]
  int b = blockIdx.z;
  int t = threadIdx.x;                 // 0..127
  int w = t >> 5, lane = t & 31;
  int m0 = blockIdx.x * 64;
  int n0 = blockIdx.y * 64;
  const f16* Ab = A  + (long)b * sA + (long)m0 * lda;
  const f16* Bb = Bm + (long)b * sB + (long)n0 * ldb;

  // staging assignment: thread t covers row (t>>1), two 16B segments.
  int srow = t >> 1;
  int sseg = (t & 1) * 2;              // segment index (16B units)
  unsigned ldsA[2], ldsB[2];
#pragma unroll
  for (int u = 0; u < 2; ++u) {
    ldsA[u] = (unsigned)(size_t)&sbuf[u][0][srow * 32 + sseg * 8];
    ldsB[u] = (unsigned)(size_t)&sbuf[u][1][srow * 32 + sseg * 8];
  }
  const f16* gaRow = Ab + (long)srow * lda + sseg * 8;
  const f16* gbRow = Bb + (long)srow * ldb + sseg * 8;

  auto stage = [&](int u, int k) {
    async_cp16(ldsA[u],      gaRow + k);
    async_cp16(ldsA[u] + 16, gaRow + k + 8);
    async_cp16(ldsB[u],      gbRow + k);
    async_cp16(ldsB[u] + 16, gbRow + k + 8);
  };

  stage(0, 0);
  v8f a00 = {}, a01 = {}, a10 = {}, a11 = {};
  int wm = (w & 1) * 32;               // wave offset in m within block tile
  int wn = (w >> 1) * 32;              // wave offset in n
  int steps = K / 32;
  for (int i = 0; i < steps; ++i) {
    int cur = i & 1;
    async_wait0();                     // buffer `cur` resident in LDS
    __syncthreads();                   // visible to all waves; old reads done
    if (i + 1 < steps) stage(cur ^ 1, (i + 1) * 32);
    const f16* la = &sbuf[cur][0][0];
    const f16* lb = &sbuf[cur][1][0];
    v16h fa0 = frag_ld(la + (wm +  0) * 32, 32);
    v16h fa1 = frag_ld(la + (wm + 16) * 32, 32);
    v16h fb0 = frag_ld(lb + (wn +  0) * 32, 32);
    v16h fb1 = frag_ld(lb + (wn + 16) * 32, 32);
    a00 = wmma16(fa0, fb0, a00);
    a01 = wmma16(fa0, fb1, a01);
    a10 = wmma16(fa1, fb0, a10);
    a11 = wmma16(fa1, fb1, a11);
  }
  int n  = lane & 15;
  int mb = (lane & 16) ? 8 : 0;
#pragma unroll
  for (int s = 0; s < 4; ++s) {
    const v8f* accs[4] = {&a00, &a01, &a10, &a11};
    v8f acc = *accs[s];
    int mo = (s >> 1) * 16, no = (s & 1) * 16;
    int col = n0 + wn + no + n;
    float bv = bias ? bias[col] : 0.0f;
#pragma unroll
    for (int r = 0; r < 8; ++r) {
      int row = m0 + wm + mo + mb + r;
      long off = (long)b * sD + (long)row * ldd + col;
      float v = acc[r] + bv;
      if (resid) v += resid[(long)b * sR + (long)row * ldd + col];
      if (Df) Df[off] = v;
      if (Dh) Dh[off] = (f16)v;
    }
  }
}

// ---------------------------------------------------------------------------
// Fused masked flash attention (one wave = one 16-row Q tile of one (b,h)).
// grid = (N/128, H, B), block = 256 (8 waves). Never materializes NxN scores.
// ---------------------------------------------------------------------------
__global__ void k_attn_flash(const f16* __restrict__ Qh,   // [B,N,C]
                             const f16* __restrict__ KVh,  // [B,N,2C] (K|V)
                             const f16* __restrict__ Vt,   // [B,H,HD,N]
                             const int* __restrict__ Am,   // [N,N]
                             f16* __restrict__ Oh)         // [B,N,C]
{
  __shared__ __align__(16) f16 pstage[8][16 * 32];
  int w    = threadIdx.x >> 5;
  int lane = threadIdx.x & 31;
  int h = blockIdx.y, b = blockIdx.z;
  int q0 = blockIdx.x * 128 + w * 16;
  const float scale = 0.17677669529663687f; // hd^-0.5, hd=32

  v16h aq = frag_ld(Qh + ((long)(b * N_ + q0)) * C_ + h * HD_, C_);
  v8f o0 = {}, o1 = {};
  float mi[8], li[8];
#pragma unroll
  for (int r = 0; r < 8; ++r) { mi[r] = -1e30f; li[r] = 0.0f; }
  int n  = lane & 15;
  int mb = (lane & 16) ? 8 : 0;
  f16* st = pstage[w];

  for (int c0 = 0; c0 < N_; c0 += 32) {
    if (c0 + 128 < N_)  // stream the mask ahead (global_prefetch_b8)
      __builtin_prefetch(Am + (long)(q0 + mb) * N_ + c0 + 128, 0, 1);
    v16h bk0 = frag_ld(KVh + ((long)(b * N_ + c0)) * (2 * C_) + h * HD_, 2 * C_);
    v16h bk1 = frag_ld(KVh + ((long)(b * N_ + c0 + 16)) * (2 * C_) + h * HD_, 2 * C_);
    v8f z = {};
    v8f s0 = wmma16(aq, bk0, z);
    v8f s1 = wmma16(aq, bk1, z);
#pragma unroll
    for (int r = 0; r < 8; ++r) {
      int row = q0 + mb + r;
      float v0 = s0[r] * scale, v1 = s1[r] * scale;
      v0 = (Am[(long)row * N_ + c0 + n]      > 0) ? v0 : -9e15f;
      v1 = (Am[(long)row * N_ + c0 + 16 + n] > 0) ? v1 : -9e15f;
      float rm = fmaxf(v0, v1);
#pragma unroll
      for (int d = 1; d < 16; d <<= 1) rm = fmaxf(rm, __shfl_xor(rm, d, 32));
      float mnew = fmaxf(mi[r], rm);
      float fsc  = __expf(mi[r] - mnew);
      float p0   = __expf(v0 - mnew);
      float p1   = __expf(v1 - mnew);
      float ls   = p0 + p1;
#pragma unroll
      for (int d = 1; d < 16; d <<= 1) ls += __shfl_xor(ls, d, 32);
      li[r] = li[r] * fsc + ls;
      mi[r] = mnew;
      o0[r] *= fsc; o1[r] *= fsc;
      st[(mb + r) * 32 + n]      = (f16)p0;   // stage P in A-layout source
      st[(mb + r) * 32 + 16 + n] = (f16)p1;
    }
    v16h ap  = frag_ld(st, 32);
    v16h bv0 = frag_ld(Vt + ((long)((b * H_ + h) * HD_ + 0))  * N_ + c0, N_);
    v16h bv1 = frag_ld(Vt + ((long)((b * H_ + h) * HD_ + 16)) * N_ + c0, N_);
    o0 = wmma16(ap, bv0, o0);
    o1 = wmma16(ap, bv1, o1);
  }
#pragma unroll
  for (int r = 0; r < 8; ++r) {
    float inv = 1.0f / li[r];
    int row = q0 + mb + r;
    long base = ((long)(b * N_ + row)) * C_ + h * HD_;
    Oh[base + n]      = (f16)(o0[r] * inv);
    Oh[base + 16 + n] = (f16)(o1[r] * inv);
  }
}

// ---------------------------------------------------------------------------
// Per-row: softmax -> top-k(kk) threshold (binary search on prob in [0,1])
// -> masked re-softmax -> f16 out.  grid=(N, B), block=256, 8 elems/thread.
// ---------------------------------------------------------------------------
__device__ __forceinline__ float blk_red_max(float v, float* red) {
  int t = threadIdx.x; red[t] = v; __syncthreads();
  for (int s = 128; s > 0; s >>= 1) { if (t < s) red[t] = fmaxf(red[t], red[t + s]); __syncthreads(); }
  float r = red[0]; __syncthreads(); return r;
}
__device__ __forceinline__ float blk_red_sum(float v, float* red) {
  int t = threadIdx.x; red[t] = v; __syncthreads();
  for (int s = 128; s > 0; s >>= 1) { if (t < s) red[t] += red[t + s]; __syncthreads(); }
  float r = red[0]; __syncthreads(); return r;
}

__global__ void k_softmax_topk(const float* __restrict__ logits,
                               f16* __restrict__ out, int kk)
{
  __shared__ float red[256];
  int row = blockIdx.x, b = blockIdx.y, t = threadIdx.x;
  const float* p = logits + ((long)b * N_ + row) * N_;
  f16* o = out + ((long)b * N_ + row) * N_;
  float v[8];
#pragma unroll
  for (int i = 0; i < 8; ++i) v[i] = p[t + i * 256];
  float mx = -1e30f;
#pragma unroll
  for (int i = 0; i < 8; ++i) mx = fmaxf(mx, v[i]);
  mx = blk_red_max(mx, red);
  float sm = 0.0f;
#pragma unroll
  for (int i = 0; i < 8; ++i) { v[i] = __expf(v[i] - mx); sm += v[i]; }
  sm = blk_red_sum(sm, red);
  float inv = 1.0f / sm;
#pragma unroll
  for (int i = 0; i < 8; ++i) v[i] *= inv;       // prob in (0,1]
  float lo = 0.0f, hi = 1.0f;
  for (int it = 0; it < 30; ++it) {
    float mid = 0.5f * (lo + hi);
    float c = 0.0f;
#pragma unroll
    for (int i = 0; i < 8; ++i) c += (v[i] >= mid) ? 1.0f : 0.0f;
    c = blk_red_sum(c, red);
    if (c >= (float)kk) lo = mid; else hi = mid;
  }
  float thr = lo;
  float mx2 = -1e30f;
#pragma unroll
  for (int i = 0; i < 8; ++i) if (v[i] >= thr) mx2 = fmaxf(mx2, v[i]);
  mx2 = blk_red_max(mx2, red);
  float s2 = 0.0f;
  float e[8];
#pragma unroll
  for (int i = 0; i < 8; ++i) { e[i] = (v[i] >= thr) ? __expf(v[i] - mx2) : 0.0f; s2 += e[i]; }
  s2 = blk_red_sum(s2, red);
  float inv2 = 1.0f / s2;
#pragma unroll
  for (int i = 0; i < 8; ++i) o[t + i * 256] = (f16)(e[i] * inv2);
}

// --------------------------- small utility kernels -------------------------
__global__ void k_cvt(const float* __restrict__ s, f16* __restrict__ d, long n) {
  long i = (long)blockIdx.x * blockDim.x + threadIdx.x;
  long st = (long)gridDim.x * blockDim.x;
  for (; i < n; i += st) d[i] = (f16)s[i];
}

// Vt[b,h,d,n] = KV[b,n, C + h*HD + d]
__global__ void k_build_vt(const f16* __restrict__ KVh, f16* __restrict__ Vt) {
  long total = (long)B_ * H_ * HD_ * N_;
  long i = (long)blockIdx.x * blockDim.x + threadIdx.x;
  long st = (long)gridDim.x * blockDim.x;
  for (; i < total; i += st) {
    int n = (int)(i % N_); long q = i / N_;
    int d = (int)(q % HD_); q /= HD_;
    int h = (int)(q % H_);  int b = (int)(q / H_);
    Vt[i] = KVh[((long)(b * N_ + n)) * (2 * C_) + C_ + h * HD_ + d];
  }
}

// gT[c,i] = g[i,c]  (CxC, f32 -> f16)
__global__ void k_transpose_cvt_w(const float* __restrict__ g, f16* __restrict__ gT) {
  long total = (long)C_ * C_;
  long idx = (long)blockIdx.x * blockDim.x + threadIdx.x;
  long st = (long)gridDim.x * blockDim.x;
  for (; idx < total; idx += st) {
    int i = (int)(idx % C_); int c = (int)(idx / C_);
    gT[(long)c * C_ + i] = (f16)g[(long)i * C_ + c];
  }
}

// d[b,c,n] = s[b,n,c]   ([B,N,C] -> [B,C,N], f16)
__global__ void k_transpose_bnc(const f16* __restrict__ s, f16* __restrict__ d) {
  long total = (long)B_ * N_ * C_;
  long i = (long)blockIdx.x * blockDim.x + threadIdx.x;
  long st = (long)gridDim.x * blockDim.x;
  for (; i < total; i += st) {
    int n = (int)(i % N_); long q = i / N_;
    int c = (int)(q % C_); int b = (int)(q / C_);
    d[i] = s[((long)(b * N_ + n)) * C_ + c];
  }
}

// ---------------------------------------------------------------------------
extern "C" void kernel_launch(void* const* d_in, const int* in_sizes, int n_in,
                              void* d_out, int out_size, void* d_ws, size_t ws_size,
                              hipStream_t stream) {
  (void)in_sizes; (void)n_in; (void)out_size; (void)ws_size;
  const float* x0    = (const float*)d_in[0];
  const int*   Am    = (const int*)  d_in[1];
  const float* qw[2]  = {(const float*)d_in[2], (const float*)d_in[6]};
  const float* kvw[2] = {(const float*)d_in[3], (const float*)d_in[7]};
  const float* pw[2]  = {(const float*)d_in[4], (const float*)d_in[8]};
  const float* pb[2]  = {(const float*)d_in[5], (const float*)d_in[9]};
  const float* cq_w = (const float*)d_in[10]; const float* cq_b = (const float*)d_in[11];
  const float* ck_w = (const float*)d_in[12]; const float* ck_b = (const float*)d_in[13];
  const float* cv_w = (const float*)d_in[14]; const float* cv_b = (const float*)d_in[15];
  const float* gc_w = (const float*)d_in[16]; const float* gc_b = (const float*)d_in[17];

  const long BNC = (long)B_ * N_ * C_;      // 524288
  const long BNN = (long)B_ * N_ * N_;      // 8388608
  char* ws = (char*)d_ws; size_t off = 0;
  auto alloc = [&](size_t bytes) -> void* {
    void* p = ws + off; off = (off + bytes + 255) & ~(size_t)255; return p;
  };
  f16* xh    = (f16*)alloc(BNC * 2);
  f16* wqh[2]  = {(f16*)alloc(C_*C_*2),   (f16*)alloc(C_*C_*2)};
  f16* wkvh[2] = {(f16*)alloc(2*C_*C_*2), (f16*)alloc(2*C_*C_*2)};
  f16* wph[2]  = {(f16*)alloc(C_*C_*2),   (f16*)alloc(C_*C_*2)};
  f16* cqh   = (f16*)alloc(C4_*C_*2);
  f16* ckh   = (f16*)alloc(C4_*C_*2);
  f16* cvh   = (f16*)alloc(C_*C_*2);
  f16* gcTh  = (f16*)alloc(C_*C_*2);
  f16* Qh    = (f16*)alloc(BNC * 2);
  f16* KVh   = (f16*)alloc(BNC * 4);
  f16* Vth   = (f16*)alloc(BNC * 2);
  f16* AOh   = (f16*)alloc(BNC * 2);
  float* x1f = (float*)alloc(BNC * 4);
  f16*   x1h = (f16*)alloc(BNC * 2);
  float* x2f = (float*)alloc(BNC * 4);
  f16*   x2h = (f16*)alloc(BNC * 2);
  f16* Qgh   = (f16*)alloc((long)B_*N_*C4_*2);
  f16* Kgh   = (f16*)alloc((long)B_*N_*C4_*2);
  float* adjf= (float*)alloc(BNN * 4);
  f16*  adjh = (f16*)alloc(BNN * 2);
  f16*  Valh = (f16*)alloc(BNC * 2);
  f16*  Suph = (f16*)alloc(BNC * 2);
  f16*  SupT = (f16*)alloc(BNC * 2);

  auto cvt = [&](const float* s, f16* d, long n) {
    int g = (int)((n + 255) / 256); if (g > 4096) g = 4096;
    k_cvt<<<g, 256, 0, stream>>>(s, d, n);
  };
  cvt(x0, xh, BNC);
  for (int i = 0; i < 2; ++i) {
    cvt(qw[i],  wqh[i],  (long)C_*C_);
    cvt(kvw[i], wkvh[i], (long)2*C_*C_);
    cvt(pw[i],  wph[i],  (long)C_*C_);
  }
  cvt(cq_w, cqh, (long)C4_*C_);
  cvt(ck_w, ckh, (long)C4_*C_);
  cvt(cv_w, cvh, (long)C_*C_);
  k_transpose_cvt_w<<<64, 256, 0, stream>>>(gc_w, gcTh);

  // async-staged GEMM (requires M%64==0, Nn%64==0, K%32==0)
  auto gemmA = [&](const f16* A, int lda, long sA, const f16* Bm, int ldb, long sB,
                   const float* bias, const float* resid, long sR,
                   float* Df, f16* Dh, int ldd, long sD, int M, int Nn, int K) {
    dim3 grid(M / 64, Nn / 64, B_);
    k_gemm_nt_async<<<grid, 128, 0, stream>>>(A, lda, sA, Bm, ldb, sB, bias,
                                              resid, sR, Df, Dh, ldd, sD, K);
  };
  // plain GEMM with wave arrangement wm x wn (wm*wn==4)
  auto gemmP = [&](const f16* A, int lda, long sA, const f16* Bm, int ldb, long sB,
                   const float* bias, const float* resid, long sR,
                   float* Df, f16* Dh, int ldd, long sD, int M, int Nn, int K,
                   int wm, int wn) {
    dim3 grid(M / (32 * wm), Nn / (32 * wn), B_);
    k_gemm_nt<<<grid, 128, 0, stream>>>(A, lda, sA, Bm, ldb, sB, bias,
                                        resid, sR, Df, Dh, ldd, sD, K, wm, wn);
  };

  // --- two attention blocks ---
  const f16*   xin_h = xh;
  const float* xin_f = x0;
  float* xof[2] = {x1f, x2f};
  f16*   xoh[2] = {x1h, x2h};
  for (int blk = 0; blk < 2; ++blk) {
    // Q = x @ qw.T            [B,N,C]
    gemmA(xin_h, C_, (long)N_*C_, wqh[blk], C_, 0, nullptr, nullptr, 0,
          nullptr, Qh, C_, (long)N_*C_, N_, C_, C_);
    // KV = x @ kv_w.T         [B,N,2C]
    gemmA(xin_h, C_, (long)N_*C_, wkvh[blk], C_, 0, nullptr, nullptr, 0,
          nullptr, KVh, 2*C_, (long)N_*2*C_, N_, 2*C_, C_);
    k_build_vt<<<2048, 256, 0, stream>>>(KVh, Vth);
    dim3 ga(N_ / 128, H_, B_);
    k_attn_flash<<<ga, 256, 0, stream>>>(Qh, KVh, Vth, Am, AOh);
    // x = attn_out @ pw.T + pb + x
    gemmA(AOh, C_, (long)N_*C_, wph[blk], C_, 0, pb[blk], xin_f, (long)N_*C_,
          xof[blk], xoh[blk], C_, (long)N_*C_, N_, C_, C_);
    xin_h = xoh[blk]; xin_f = xof[blk];
  }

  // --- graph convolution ---
  // query/key (conv1d k=1 == GEMM), [B,N,C4]  (Nn=32 -> plain, 4x1 waves)
  gemmP(x2h, C_, (long)N_*C_, cqh, C_, 0, cq_b, nullptr, 0,
        nullptr, Qgh, C4_, (long)N_*C4_, N_, C4_, C_, 4, 1);
  gemmP(x2h, C_, (long)N_*C_, ckh, C_, 0, ck_b, nullptr, 0,
        nullptr, Kgh, C4_, (long)N_*C4_, N_, C4_, C_, 4, 1);
  // adj logits [B,N,N] = Kg @ Qg.T  (K = C4 = 32, single k-step -> plain)
  gemmP(Kgh, C4_, (long)N_*C4_, Qgh, C4_, (long)N_*C4_, nullptr, nullptr, 0,
        adjf, nullptr, N_, (long)N_*N_, N_, N_, C4_, 2, 2);
  // softmax -> top-k(2/3) -> masked softmax -> f16 adjacency
  int kk = (N_ / 3) * 2;  // (2048/3)*2 = 1364
  dim3 gt(N_, B_);
  k_softmax_topk<<<gt, 256, 0, stream>>>(adjf, adjh, kk);
  // value [B,N,C], support = value @ gc_w (via gc_w^T for NT form)
  gemmA(x2h, C_, (long)N_*C_, cvh, C_, 0, cv_b, nullptr, 0,
        nullptr, Valh, C_, (long)N_*C_, N_, C_, C_);
  gemmA(Valh, C_, (long)N_*C_, gcTh, C_, 0, nullptr, nullptr, 0,
        nullptr, Suph, C_, (long)N_*C_, N_, C_, C_);
  k_transpose_bnc<<<2048, 256, 0, stream>>>(Suph, SupT);
  // out = adj @ support + gc_b   (K = N = 2048) -> f32 d_out
  gemmA(adjh, N_, (long)N_*N_, SupT, N_, (long)C_*N_, gc_b, nullptr, 0,
        (float*)d_out, nullptr, C_, (long)N_*C_, N_, C_, N_);
}
